// ClusterMemory_40450001994250
// MI455X (gfx1250) — matrix operand order repaired and checked by reading, not compiled, not run on hardware
//
#include <hip/hip_runtime.h>
#include <hip/hip_bf16.h>

#define TEMP_ 0.05f
#define MOM_  0.2f
#define B_    256
#define D_    768
#define NI_   65536
#define NC_   8192
#define T_    128
#define K_    50
#define RATE_ 38

typedef __attribute__((ext_vector_type(16))) __bf16 v16bf;
typedef __attribute__((ext_vector_type(8)))  float  v8f;

__device__ __forceinline__ unsigned short f32_to_bf16(float f) {
    unsigned int u = __builtin_bit_cast(unsigned int, f);
    unsigned int r = u + 0x7fffu + ((u >> 16) & 1u);   // round-to-nearest-even
    return (unsigned short)(r >> 16);
}
__device__ __forceinline__ unsigned int pack_bf16(float a, float b) {
    return (unsigned int)f32_to_bf16(a) | ((unsigned int)f32_to_bf16(b) << 16);
}

// ---------------------------------------------------------------- utilities
__global__ void k_init(float* sc) {
    sc[0] = 0.f; sc[1] = 0.f; sc[2] = 0.f;
}

__global__ void k_combine(const float* sc, float* out) {
    out[0] = sc[0] + sc[1] + sc[2];
}

// ---------------------------------------------------------------- x = l2norm(inputs)
__global__ __launch_bounds__(256) void k_l2norm(const float* __restrict__ in,
                                                float* __restrict__ x) {
    __shared__ float red[256];
    int b = blockIdx.x, t = threadIdx.x;
    float s = 0.f;
    for (int d = t; d < D_; d += 256) { float v = in[(size_t)b * D_ + d]; s += v * v; }
    red[t] = s; __syncthreads();
    for (int st = 128; st > 0; st >>= 1) { if (t < st) red[t] += red[t + st]; __syncthreads(); }
    float rn = 1.0f / sqrtf(red[0]);
    for (int d = t; d < D_; d += 256) x[(size_t)b * D_ + d] = in[(size_t)b * D_ + d] * rn;
}

// ---------------------------------------------------------------- patch loss
__global__ __launch_bounds__(256) void k_patch(const float* __restrict__ cls,
                                               const float* __restrict__ part,
                                               const float* __restrict__ tok,
                                               float* __restrict__ acc) {
    __shared__ float q[D_];
    __shared__ float vals[T_];
    __shared__ float red[256];
    __shared__ float maxv_s;
    int b = blockIdx.x, t = threadIdx.x;
    for (int d = t; d < D_; d += 256)
        q[d] = 0.5f * (cls[(size_t)b * D_ + d] + part[(size_t)b * D_ + d]);
    __syncthreads();
    float v = 0.f;
    if (t < T_) {
        const float* tp = tok + ((size_t)b * T_ + t) * D_;
        float s = 0.f;
        for (int d = 0; d < D_; ++d) s += q[d] * tp[d];
        vals[t] = s; v = s;
    }
    __syncthreads();
    int rank = 0;
    if (t < T_) {
        for (int u = 0; u < T_; ++u) {
            float w = vals[u];
            rank += (w < v) || (w == v && u < t);
        }
        if (rank == T_ - 1) maxv_s = v;   // row max == logits[0]
    }
    __syncthreads();
    float e = 0.f;
    if (t < T_ && (rank == T_ - 1 || rank < RATE_))
        e = __expf((v - maxv_s) / TEMP_);
    red[t] = e; __syncthreads();
    for (int st = 128; st > 0; st >>= 1) { if (t < st) red[t] += red[t + st]; __syncthreads(); }
    if (t == 0) atomicAdd(acc, __logf(red[0]) / (float)B_);
}

// ---------------------------------------------------------------- bf16 WMMA GEMM: out[B_,N] = X[B_,D_] * W[N,D_]^T
// 256x64 C tile (full batch in one row-block -> W streamed exactly once),
// TK=64 slab, 4x2 wave grid (each wave: 64x32 = 4x2 subtiles).
// LDS row stride 72 shorts (144 B): 16B-aligned fragments, 9*ln mod 64 bank walk.
// Optionally mirrors the streamed W tile into `mirror` (fused new_mem copy).
#define TM_ 256
#define TN_ 64
#define TK_ 64
#define LDSS_ 72

__global__ __launch_bounds__(256) void k_gemm_xt(const float* __restrict__ X,
                                                 const float* __restrict__ W,
                                                 float* __restrict__ out, int N,
                                                 float* __restrict__ mirror) {
    __shared__ __attribute__((aligned(16))) unsigned short a_lds[TM_ * LDSS_];
    __shared__ __attribute__((aligned(16))) unsigned short b_lds[TN_ * LDSS_];

    const int tid  = threadIdx.x;
    const int lane = tid & 31;
    const int wv   = tid >> 5;
    const int wr   = wv >> 1;          // 0..3 : 64-row strip
    const int wc   = wv & 1;           // 0..1 : 32-col strip
    const int h    = lane >> 4;
    const int ln   = lane & 15;
    const int col0 = blockIdx.x * TN_;

    v8f acc[4][2] = {};

    for (int kk = 0; kk < D_; kk += TK_) {
        // cooperative load: float4 global -> 4 packed bf16 -> ds_store_b64
        for (int q = tid; q < TM_ * (TK_ / 4); q += 256) {
            int r = q >> 4, k4 = (q & 15) << 2;
            float4 f = *(const float4*)&X[(size_t)r * D_ + kk + k4];
            *(uint2*)&a_lds[r * LDSS_ + k4] =
                make_uint2(pack_bf16(f.x, f.y), pack_bf16(f.z, f.w));
        }
        for (int q = tid; q < TN_ * (TK_ / 4); q += 256) {
            int c = q >> 4, k4 = (q & 15) << 2;
            size_t off = (size_t)(col0 + c) * D_ + kk + k4;
            float4 f = *(const float4*)&W[off];
            if (mirror) {                      // fused new_mem copy (4B-aligned dst)
                float* mp = mirror + off;
                mp[0] = f.x; mp[1] = f.y; mp[2] = f.z; mp[3] = f.w;
            }
            if (kk + TK_ < D_)
                __builtin_prefetch(&W[off + TK_], 0, 1);
            *(uint2*)&b_lds[c * LDSS_ + k4] =
                make_uint2(pack_bf16(f.x, f.y), pack_bf16(f.z, f.w));
        }
        __syncthreads();

#pragma unroll
        for (int k2 = 0; k2 < TK_; k2 += 32) {
            // A fragments: lane row m = ln; K runs [k2+h*8, +8) and [k2+16+h*8, +8)
            union { v16bf v; uint4 q[2]; } af[4];
#pragma unroll
            for (int mi = 0; mi < 4; ++mi) {
                const unsigned short* arow =
                    &a_lds[(wr * 64 + mi * 16 + ln) * LDSS_ + k2 + h * 8];
                af[mi].q[0] = *(const uint4*)(arow);
                af[mi].q[1] = *(const uint4*)(arow + 16);
            }
#pragma unroll
            for (int ni = 0; ni < 2; ++ni) {
                // B fragment: lane col n = ln; K runs [k2+h*8, +8) and [k2+16+h*8, +8)
                union { v16bf v; uint4 q[2]; } bfr;
                const unsigned short* brow =
                    &b_lds[(wc * 32 + ni * 16 + ln) * LDSS_ + k2 + h * 8];
                bfr.q[0] = *(const uint4*)(brow);
                bfr.q[1] = *(const uint4*)(brow + 16);
#pragma unroll
                for (int mi = 0; mi < 4; ++mi)
                    acc[mi][ni] = __builtin_amdgcn_wmma_f32_16x16x32_bf16(
                        false, af[mi].v, false, bfr.v, (short)0, acc[mi][ni], false, false);
            }
        }
        __syncthreads();
    }

    // C/D layout: VGPR r holds (m = r + 8*h, n = ln)
#pragma unroll
    for (int mi = 0; mi < 4; ++mi)
#pragma unroll
        for (int ni = 0; ni < 2; ++ni)
#pragma unroll
            for (int r = 0; r < 8; ++r) {
                int m = wr * 64 + mi * 16 + r + 8 * h;
                int n = wc * 32 + ni * 16 + ln;
                out[(size_t)m * N + col0 + n] = acc[mi][ni][r];
            }
}

// ---------------------------------------------------------------- anchor loss (mutates mat scratch in place)
__global__ __launch_bounds__(256) void k_anchor(float* __restrict__ mat,
                                                const int* __restrict__ mem_labels,
                                                const int* __restrict__ targets,
                                                float* __restrict__ acc) {
    __shared__ float cmax[256]; __shared__ int carg[256];
    __shared__ float red[256];  __shared__ int redi[256];
    __shared__ float topk[K_];
    __shared__ float sposmin;   __shared__ int ssel;

    int b = blockIdx.x, t = threadIdx.x;
    int tgt = targets[b];
    float* row = mat + (size_t)b * NI_;

    const float NINF = -__builtin_inff();
    float pmin = __builtin_inff();
    float lmax = NINF; int larg = 0;
    for (int i = 0; i < NI_ / 256; ++i) {
        int j = i * 256 + t;                 // coalesced; thread t owns strided chunk j%256==t
        float v = row[j];
        if (mem_labels[j] == tgt) { pmin = fminf(pmin, v); v = NINF; }
        row[j] = v;                          // masked copy back into scratch
        if (v > lmax) { lmax = v; larg = j; }
    }
    cmax[t] = lmax; carg[t] = larg;
    red[t]  = pmin;
    __syncthreads();
    for (int st = 128; st > 0; st >>= 1) { if (t < st) red[t] = fminf(red[t], red[t + st]); __syncthreads(); }
    if (t == 0) sposmin = red[0];

    for (int it = 0; it < K_; ++it) {
        __syncthreads();
        red[t] = cmax[t]; redi[t] = carg[t];
        __syncthreads();
        for (int st = 128; st > 0; st >>= 1) {
            if (t < st && red[t + st] > red[t]) { red[t] = red[t + st]; redi[t] = redi[t + st]; }
            __syncthreads();
        }
        if (t == 0) { topk[it] = red[0]; ssel = redi[0]; }
        __syncthreads();
        int sj = ssel;
        if ((sj & 255) == t) {               // owner rescans its strided chunk
            row[sj] = NINF;
            float m2 = NINF; int a2 = 0;
            for (int i = 0; i < NI_ / 256; ++i) {
                int j = i * 256 + t;
                float v = row[j];
                if (v > m2) { m2 = v; a2 = j; }
            }
            cmax[t] = m2; carg[t] = a2;
        }
    }
    __syncthreads();
    if (t == 0) {
        float l0 = sposmin / TEMP_;
        float m  = fmaxf(l0, topk[0] / TEMP_);
        float s  = __expf(l0 - m);
        for (int i = 0; i < K_; ++i) s += __expf(topk[i] / TEMP_ - m);
        atomicAdd(acc, (m + __logf(s) - l0) / (float)B_);
    }
}

// ---------------------------------------------------------------- cluster loss
__global__ __launch_bounds__(256) void k_cluster(const float* __restrict__ lg,
                                                 const int* __restrict__ targets,
                                                 float* __restrict__ acc) {
    __shared__ float red[256];
    int b = blockIdx.x, t = threadIdx.x;
    const float* row = lg + (size_t)b * NC_;
    float m = -__builtin_inff();
    for (int j = t; j < NC_; j += 256) m = fmaxf(m, row[j]);
    red[t] = m; __syncthreads();
    for (int st = 128; st > 0; st >>= 1) { if (t < st) red[t] = fmaxf(red[t], red[t + st]); __syncthreads(); }
    m = red[0] / TEMP_;
    __syncthreads();
    float s = 0.f;
    for (int j = t; j < NC_; j += 256) s += __expf(row[j] / TEMP_ - m);
    red[t] = s; __syncthreads();
    for (int st = 128; st > 0; st >>= 1) { if (t < st) red[t] += red[t + st]; __syncthreads(); }
    if (t == 0) {
        float lse = m + __logf(red[0]);
        float tl  = row[targets[b]] / TEMP_;
        atomicAdd(acc, (lse - tl) / (float)B_);
    }
}

// ---------------------------------------------------------------- scatter momentum update (after fused copy)
__global__ __launch_bounds__(256) void k_scatter(const float* __restrict__ memf,
                                                 const float* __restrict__ x,
                                                 const int* __restrict__ indexes,
                                                 float* __restrict__ new_mem) {
    __shared__ float red[256];
    __shared__ float upd[D_];
    int b = blockIdx.x, t = threadIdx.x;
    int idx = indexes[b];
    float s = 0.f;
    for (int d = t; d < D_; d += 256) {
        float u = MOM_ * memf[(size_t)idx * D_ + d] + (1.0f - MOM_) * x[(size_t)b * D_ + d];
        upd[d] = u; s += u * u;
    }
    red[t] = s; __syncthreads();
    for (int st = 128; st > 0; st >>= 1) { if (t < st) red[t] += red[t + st]; __syncthreads(); }
    float rn = 1.0f / sqrtf(red[0]);
    for (int d = t; d < D_; d += 256) new_mem[(size_t)idx * D_ + d] = upd[d] * rn;
}

// ---------------------------------------------------------------- launch
extern "C" void kernel_launch(void* const* d_in, const int* in_sizes, int n_in,
                              void* d_out, int out_size, void* d_ws, size_t ws_size,
                              hipStream_t stream) {
    const float* inputs = (const float*)d_in[0];
    const float* cls    = (const float*)d_in[1];
    const float* part   = (const float*)d_in[2];
    const float* tokens = (const float*)d_in[3];
    const float* memf   = (const float*)d_in[4];
    const float* clf    = (const float*)d_in[5];
    const int*   targets = (const int*)d_in[6];
    const int*   indexes = (const int*)d_in[7];
    const int*   mlabels = (const int*)d_in[8];

    float* out     = (float*)d_out;
    float* new_mem = out + 1;                       // [NI_, D_]

    float* mat = (float*)d_ws;                      // B_*NI_  (64 MB, fits L2)
    float* clg = mat + (size_t)B_ * NI_;            // B_*NC_  (8 MB)
    float* x   = clg + (size_t)B_ * NC_;            // B_*D_
    float* sc  = x + (size_t)B_ * D_;               // 3 loss scalars

    k_init<<<1, 1, 0, stream>>>(sc);
    k_l2norm<<<B_, 256, 0, stream>>>(inputs, x);
    k_patch<<<B_, 256, 0, stream>>>(cls, part, tokens, sc + 0);

    // mem GEMM streams mem exactly once and mirrors it into new_mem (fused copy)
    k_gemm_xt<<<dim3(NI_ / TN_), 256, 0, stream>>>(x, memf, mat, NI_, new_mem);
    k_gemm_xt<<<dim3(NC_ / TN_), 256, 0, stream>>>(x, clf, clg, NC_, nullptr);

    k_anchor<<<B_, 256, 0, stream>>>(mat, mlabels, targets, sc + 1);
    k_cluster<<<B_, 256, 0, stream>>>(clg, targets, sc + 2);
    k_combine<<<1, 1, 0, stream>>>(sc, out);

    k_scatter<<<B_, 256, 0, stream>>>(memf, x, indexes, new_mem);
}
